// TemporalFusionTransformer_8504035246312
// MI455X (gfx1250) — matrix operand order repaired and checked by reading, not compiled, not run on hardware
//
#include <hip/hip_runtime.h>
#include <math.h>

typedef __bf16 bf16;
typedef __attribute__((ext_vector_type(16))) __bf16 v16bf;
typedef __attribute__((ext_vector_type(8)))  __bf16 v8bf;
typedef __attribute__((ext_vector_type(8)))  float  v8f;

#define D_MODEL 256
#define NTOK    8192   // B*T = 8*1024
#define TSEQ    1024
#define NHEAD   4

// ---- CDNA5 async global->LDS path (guarded: falls back to sync staging) ----
#if defined(__has_builtin)
#if __has_builtin(__builtin_amdgcn_global_load_async_to_lds_b128)
#define HAVE_ASYNC_LDS 1
#endif
#if __has_builtin(__builtin_amdgcn_s_wait_asynccnt)
#define HAVE_WAIT_ASYNC 1
#endif
#endif

// 128-bit payload type + address-space-qualified flavors for the async builtin
typedef __attribute__((__vector_size__(16))) int i32x4;
typedef __attribute__((address_space(1))) i32x4 as1_i32x4;
typedef __attribute__((address_space(3))) i32x4 as3_i32x4;

__device__ __forceinline__ void wait_async_all() {
#if defined(HAVE_WAIT_ASYNC)
  __builtin_amdgcn_s_wait_asynccnt(0);
#else
  asm volatile("s_wait_asynccnt 0x0" ::: "memory");
#endif
}

// ---------------------------------------------------------------------------
// fp32 -> bf16 weight conversion
// ---------------------------------------------------------------------------
__global__ __launch_bounds__(256) void f32_to_bf16_kernel(const float* __restrict__ src,
                                                          bf16* __restrict__ dst, int n) {
  int i = blockIdx.x * blockDim.x + threadIdx.x;
  if (i < n) dst[i] = (bf16)src[i];
}

// ---------------------------------------------------------------------------
// Generic fused WMMA GEMM: out = epilogue(A[rows,256] @ W^T + bias)
//   W is bf16 [256 out][256 in] row-major (so rows of W are WMMA-B columns).
// Block: 256 threads = 8 waves; block tile 64 rows x 256 cols.
// Wave (wm in 0..3, wn in 0..1) computes 16 rows x 128 cols = 8 C tiles.
// MODE 0: out = acc + bias            (bias may be null)
// MODE 1: out = elu(acc + bias)
// MODE 2: out = LN(resid + sigmoid(acc+bias) * gateIn)   (GRN gate + LN)
// MODE 3: out = LN(resid + acc + bias)                   (attention add + LN)
// ---------------------------------------------------------------------------
template <int MODE, typename OutT>
__global__ __launch_bounds__(256) void gemm256_kernel(
    const float* __restrict__ A, const bf16* __restrict__ W,
    const float* __restrict__ bias, const float* __restrict__ resid,
    const float* __restrict__ gateIn, const float* __restrict__ gamma,
    const float* __restrict__ beta, OutT* __restrict__ out) {
  const int lane = threadIdx.x & 31;
  const int wave = threadIdx.x >> 5;
  const int wm = wave >> 1;
  const int wn = wave & 1;
  const int l16 = lane & 15;
  const int half = lane >> 4;
  const int row0 = blockIdx.x * 64;

  v8f acc[8];
#pragma unroll
  for (int n = 0; n < 8; ++n)
#pragma unroll
    for (int i = 0; i < 8; ++i) acc[n][i] = 0.f;

  const float* arow = A + (size_t)(row0 + wm * 16 + l16) * D_MODEL;

  for (int k0 = 0; k0 < D_MODEL; k0 += 32) {
    if (k0 + 32 < D_MODEL) __builtin_prefetch(arow + k0 + 32, 0, 3);
    // ---- A fragment (16x32 bf16): elems 0..15 <- K=k0+half*16..+15
    v16bf af;
    const float* ap = arow + k0 + half * 16;
    float4 x0 = *(const float4*)(ap);
    float4 x1 = *(const float4*)(ap + 4);
    float4 x2 = *(const float4*)(ap + 8);
    float4 x3 = *(const float4*)(ap + 12);
    af[0] = (bf16)x0.x;  af[1] = (bf16)x0.y;  af[2] = (bf16)x0.z;  af[3] = (bf16)x0.w;
    af[4] = (bf16)x1.x;  af[5] = (bf16)x1.y;  af[6] = (bf16)x1.z;  af[7] = (bf16)x1.w;
    af[8] = (bf16)x2.x;  af[9] = (bf16)x2.y;  af[10] = (bf16)x2.z; af[11] = (bf16)x2.w;
    af[12] = (bf16)x3.x; af[13] = (bf16)x3.y; af[14] = (bf16)x3.z; af[15] = (bf16)x3.w;

#pragma unroll
    for (int n = 0; n < 8; ++n) {
      const int col = wn * 128 + n * 16 + l16;
      const bf16* wp = W + (size_t)col * D_MODEL + k0 + half * 16;
      v16bf bfr = *(const v16bf*)wp;  // 16 contiguous bf16: K=k0+half*16..+15
      acc[n] = __builtin_amdgcn_wmma_f32_16x16x32_bf16(false, af, false, bfr,
                                                       (short)0, acc[n], false, false);
    }
  }

  // ---- epilogue (C layout: lane holds row g+8*half, col = tile*16 + l16)
#pragma unroll
  for (int n = 0; n < 8; ++n) {
    const int col = wn * 128 + n * 16 + l16;
    const float bn = bias ? bias[col] : 0.f;
#pragma unroll
    for (int g = 0; g < 8; ++g) {
      const int r = row0 + wm * 16 + g + 8 * half;
      float v = acc[n][g] + bn;
      if constexpr (MODE == 1) {
        v = v > 0.f ? v : expm1f(v);
      } else if constexpr (MODE == 2) {
        const float e2 = gateIn[(size_t)r * D_MODEL + col];
        v = resid[(size_t)r * D_MODEL + col] + e2 / (1.f + __expf(-v));
      } else if constexpr (MODE == 3) {
        v = resid[(size_t)r * D_MODEL + col] + v;
      }
      acc[n][g] = v;
      if constexpr (MODE == 0 || MODE == 1) {
        out[(size_t)r * D_MODEL + col] = (OutT)v;
      }
    }
  }

  if constexpr (MODE >= 2) {
    // Deterministic fused LayerNorm over each full 256-wide row of the block.
    __shared__ float psum[64][33];
    __shared__ float psq[64][33];
    __shared__ float s_mean[64];
    __shared__ float s_rstd[64];

    const int slot = wn * 16 + l16;
#pragma unroll
    for (int g = 0; g < 8; ++g) {
      const int rl = wm * 16 + g + 8 * half;
      float s = 0.f, s2 = 0.f;
#pragma unroll
      for (int n = 0; n < 8; ++n) {
        const float v = acc[n][g];
        s += v;
        s2 += v * v;
      }
      psum[rl][slot] = s;
      psq[rl][slot] = s2;
    }
    __syncthreads();
    {
      const int t = threadIdx.x;
      const int rr = t >> 2, q = t & 3;
      float s = 0.f, s2 = 0.f;
#pragma unroll
      for (int i = 0; i < 8; ++i) {
        s += psum[rr][q * 8 + i];
        s2 += psq[rr][q * 8 + i];
      }
      s += __shfl_xor(s, 1);   s += __shfl_xor(s, 2);
      s2 += __shfl_xor(s2, 1); s2 += __shfl_xor(s2, 2);
      if (q == 0) {
        const float mean = s * (1.f / 256.f);
        const float var = s2 * (1.f / 256.f) - mean * mean;
        s_mean[rr] = mean;
        s_rstd[rr] = rsqrtf(var + 1e-5f);
      }
    }
    __syncthreads();
#pragma unroll
    for (int n = 0; n < 8; ++n) {
      const int col = wn * 128 + n * 16 + l16;
      const float gma = gamma[col], bta = beta[col];
#pragma unroll
      for (int g = 0; g < 8; ++g) {
        const int rl = wm * 16 + g + 8 * half;
        const float v = (acc[n][g] - s_mean[rl]) * s_rstd[rl] * gma + bta;
        out[(size_t)(row0 + rl) * D_MODEL + col] = (OutT)v;
      }
    }
  }
}

// ---------------------------------------------------------------------------
// Flash attention: grid (B*H, T/128), block 256 = 8 waves x 16 query rows.
// q/k/v: bf16 [token][256] with head h at cols h*64..h*64+63.
// Online softmax over 64-key chunks staged in LDS (V transposed for WMMA-B).
// K is staged with CDNA5 async global->LDS copies when available.
// ---------------------------------------------------------------------------
__global__ __launch_bounds__(256) void attn_kernel(const bf16* __restrict__ qb,
                                                   const bf16* __restrict__ kb,
                                                   const bf16* __restrict__ vb,
                                                   float* __restrict__ ctx) {
  __shared__ bf16 Kl[64 * 72];        // [key][dim] pitch 72
  __shared__ bf16 VT[64 * 72];        // [dim][key] pitch 72
  __shared__ bf16 Pl[8 * 16 * 72];    // per-wave P staging [16][72]

  const int lane = threadIdx.x & 31;
  const int wave = threadIdx.x >> 5;
  const int l16 = lane & 15;
  const int half = lane >> 4;
  const int b = blockIdx.x >> 2;
  const int head = blockIdx.x & (NHEAD - 1);
  const int t0 = blockIdx.y * 128 + wave * 16;
  const size_t headOff = (size_t)head * 64;

  // Resident Q fragments (16x64 -> two 16x32 A-frags)
  v16bf qf[2];
  {
    const bf16* qrow = qb + ((size_t)(b * TSEQ) + t0 + l16) * D_MODEL + headOff;
#pragma unroll
    for (int kk = 0; kk < 2; ++kk) {
      v8bf lo = *(const v8bf*)(qrow + kk * 32 + half * 8);
      v8bf hi = *(const v8bf*)(qrow + kk * 32 + 16 + half * 8);
#pragma unroll
      for (int i = 0; i < 8; ++i) {
        qf[kk][i] = lo[i];
        qf[kk][8 + i] = hi[i];
      }
    }
  }

  float mrow[8], lrow[8];
  v8f cacc[4];
#pragma unroll
  for (int g = 0; g < 8; ++g) { mrow[g] = -INFINITY; lrow[g] = 0.f; }
#pragma unroll
  for (int j = 0; j < 4; ++j)
#pragma unroll
    for (int g = 0; g < 8; ++g) cacc[j][g] = 0.f;

  const int key = threadIdx.x & 63;
  const int dgrp = threadIdx.x >> 6;  // 0..3 -> dims dgrp*16..+15

  for (int s0 = 0; s0 < TSEQ; s0 += 64) {
    __syncthreads();
    {  // cooperative staging of K (row-major) and V (transposed)
      const size_t tok = ((size_t)(b * TSEQ) + s0 + key) * D_MODEL + headOff + dgrp * 16;
#if defined(HAVE_ASYNC_LDS)
      // Direct global->LDS async copies for K (16B per lane, tracked by ASYNCcnt).
      __builtin_amdgcn_global_load_async_to_lds_b128(
          (as1_i32x4*)(kb + tok), (as3_i32x4*)(Kl + key * 72 + dgrp * 16), 0, 0);
      __builtin_amdgcn_global_load_async_to_lds_b128(
          (as1_i32x4*)(kb + tok + 8), (as3_i32x4*)(Kl + key * 72 + dgrp * 16 + 8), 0, 0);
#else
      v8bf ka = *(const v8bf*)(kb + tok);
      v8bf kbv = *(const v8bf*)(kb + tok + 8);
      *(v8bf*)(Kl + key * 72 + dgrp * 16) = ka;
      *(v8bf*)(Kl + key * 72 + dgrp * 16 + 8) = kbv;
#endif
      // V must be transposed element-wise -> manual scatter path.
      v8bf va = *(const v8bf*)(vb + tok);
      v8bf vbv = *(const v8bf*)(vb + tok + 8);
#pragma unroll
      for (int i = 0; i < 8; ++i) {
        VT[(dgrp * 16 + i) * 72 + key] = va[i];
        VT[(dgrp * 16 + 8 + i) * 72 + key] = vbv[i];
      }
#if defined(HAVE_ASYNC_LDS)
      wait_async_all();
#endif
    }
    __syncthreads();

    // S = Q K^T  (4 tiles of 16 keys, K-dim 64 -> 2 WMMA each)
    v8f sacc[4];
#pragma unroll
    for (int c = 0; c < 4; ++c)
#pragma unroll
      for (int g = 0; g < 8; ++g) sacc[c][g] = 0.f;
#pragma unroll
    for (int kk = 0; kk < 2; ++kk) {
#pragma unroll
      for (int c = 0; c < 4; ++c) {
        const bf16* kp = Kl + (c * 16 + l16) * 72 + kk * 32 + half * 8;
        v8bf klo = *(const v8bf*)kp;
        v8bf khi = *(const v8bf*)(kp + 16);
        v16bf kf;
#pragma unroll
        for (int i = 0; i < 8; ++i) { kf[i] = klo[i]; kf[8 + i] = khi[i]; }
        sacc[c] = __builtin_amdgcn_wmma_f32_16x16x32_bf16(false, qf[kk], false, kf,
                                                          (short)0, sacc[c], false, false);
      }
    }

    // online softmax update (rows reduce across the 16 lanes of each half)
    const float sc = 0.125f;  // 1/sqrt(64)
    float mnew[8], p[4][8], lsum[8];
#pragma unroll
    for (int g = 0; g < 8; ++g) mnew[g] = mrow[g];
#pragma unroll
    for (int c = 0; c < 4; ++c)
#pragma unroll
      for (int g = 0; g < 8; ++g) mnew[g] = fmaxf(mnew[g], sacc[c][g] * sc);
#pragma unroll
    for (int g = 0; g < 8; ++g) {
      mnew[g] = fmaxf(mnew[g], __shfl_xor(mnew[g], 1));
      mnew[g] = fmaxf(mnew[g], __shfl_xor(mnew[g], 2));
      mnew[g] = fmaxf(mnew[g], __shfl_xor(mnew[g], 4));
      mnew[g] = fmaxf(mnew[g], __shfl_xor(mnew[g], 8));
    }
#pragma unroll
    for (int g = 0; g < 8; ++g) {
      const float a = __expf(mrow[g] - mnew[g]);
      lrow[g] *= a;
#pragma unroll
      for (int j = 0; j < 4; ++j) cacc[j][g] *= a;
      mrow[g] = mnew[g];
      lsum[g] = 0.f;
    }
#pragma unroll
    for (int c = 0; c < 4; ++c)
#pragma unroll
      for (int g = 0; g < 8; ++g) {
        p[c][g] = __expf(sacc[c][g] * sc - mnew[g]);
        lsum[g] += p[c][g];
      }
#pragma unroll
    for (int g = 0; g < 8; ++g) {
      lsum[g] += __shfl_xor(lsum[g], 1);
      lsum[g] += __shfl_xor(lsum[g], 2);
      lsum[g] += __shfl_xor(lsum[g], 4);
      lsum[g] += __shfl_xor(lsum[g], 8);
      lrow[g] += lsum[g];
    }

    // write P (16x64) to wave-private LDS, read back as A-fragments
    bf16* pw = Pl + wave * (16 * 72);
#pragma unroll
    for (int c = 0; c < 4; ++c)
#pragma unroll
      for (int g = 0; g < 8; ++g)
        pw[(g + 8 * half) * 72 + c * 16 + l16] = (bf16)p[c][g];
    asm volatile("" ::: "memory");  // keep DS reads below DS writes (HW keeps same-wave LDS in order)

#pragma unroll
    for (int kk = 0; kk < 2; ++kk) {
      const bf16* pp = pw + l16 * 72 + kk * 32 + half * 8;
      v8bf plo = *(const v8bf*)pp;
      v8bf phi = *(const v8bf*)(pp + 16);
      v16bf pf;
#pragma unroll
      for (int i = 0; i < 8; ++i) { pf[i] = plo[i]; pf[8 + i] = phi[i]; }
#pragma unroll
      for (int j = 0; j < 4; ++j) {
        const bf16* vp = VT + (j * 16 + l16) * 72 + kk * 32 + half * 8;
        v8bf vlo = *(const v8bf*)vp;
        v8bf vhi = *(const v8bf*)(vp + 16);
        v16bf vf;
#pragma unroll
        for (int i = 0; i < 8; ++i) { vf[i] = vlo[i]; vf[8 + i] = vhi[i]; }
        cacc[j] = __builtin_amdgcn_wmma_f32_16x16x32_bf16(false, pf, false, vf,
                                                          (short)0, cacc[j], false, false);
      }
    }
  }

  // finalize: ctx = acc / l
#pragma unroll
  for (int j = 0; j < 4; ++j)
#pragma unroll
    for (int g = 0; g < 8; ++g) {
      const int trow = t0 + g + 8 * half;
      ctx[((size_t)(b * TSEQ) + trow) * D_MODEL + headOff + j * 16 + l16] =
          cacc[j][g] / lrow[g];
    }
}

// ---------------------------------------------------------------------------
// GRN2 + quantile heads, only on the last timestep of each batch (8 rows).
// Pointwise GRN + LN commute with the time slice, so this is exact.
// Single block, 256 threads, deterministic tree reductions.
// ---------------------------------------------------------------------------
__global__ __launch_bounds__(256) void grn2_head_kernel(
    const float* __restrict__ h2, const float* __restrict__ f1w,
    const float* __restrict__ f1b, const float* __restrict__ f2w,
    const float* __restrict__ f2b, const float* __restrict__ gw,
    const float* __restrict__ gb, const float* __restrict__ lng,
    const float* __restrict__ lnb, const float* __restrict__ qhw,
    const float* __restrict__ qhb, float* __restrict__ out) {
  __shared__ float xrow[256], e1[256], e2[256], red[256];
  const int t = threadIdx.x;
  for (int b = 0; b < 8; ++b) {
    const size_t base = ((size_t)b * TSEQ + (TSEQ - 1)) * D_MODEL;
    xrow[t] = h2[base + t];
    __syncthreads();
    float s = f1b[t];
    for (int k = 0; k < 256; ++k) s = fmaf(xrow[k], f1w[(size_t)t * 256 + k], s);
    e1[t] = s > 0.f ? s : expm1f(s);
    __syncthreads();
    float s2 = f2b[t];
    for (int k = 0; k < 256; ++k) s2 = fmaf(e1[k], f2w[(size_t)t * 256 + k], s2);
    e2[t] = s2;
    __syncthreads();
    float s3 = gb[t];
    for (int k = 0; k < 256; ++k) s3 = fmaf(e2[k], gw[(size_t)t * 256 + k], s3);
    const float pre = xrow[t] + e2[t] / (1.f + __expf(-s3));
    __syncthreads();
    red[t] = pre;
    __syncthreads();
    for (int st = 128; st > 0; st >>= 1) {
      if (t < st) red[t] += red[t + st];
      __syncthreads();
    }
    const float mean = red[0] * (1.f / 256.f);
    __syncthreads();
    red[t] = pre * pre;
    __syncthreads();
    for (int st = 128; st > 0; st >>= 1) {
      if (t < st) red[t] += red[t + st];
      __syncthreads();
    }
    const float var = red[0] * (1.f / 256.f) - mean * mean;
    __syncthreads();
    const float h3v = (pre - mean) * rsqrtf(var + 1e-5f) * lng[t] + lnb[t];
    for (int qi = 0; qi < 3; ++qi) {
      red[t] = h3v * qhw[qi * 256 + t];
      __syncthreads();
      for (int st = 128; st > 0; st >>= 1) {
        if (t < st) red[t] += red[t + st];
        __syncthreads();
      }
      if (t == 0) out[b * 3 + qi] = red[0] + qhb[qi];
      __syncthreads();
    }
    __syncthreads();
  }
}

// ---------------------------------------------------------------------------
extern "C" void kernel_launch(void* const* d_in, const int* in_sizes, int n_in,
                              void* d_out, int out_size, void* d_ws, size_t ws_size,
                              hipStream_t stream) {
  (void)in_sizes; (void)n_in; (void)out_size; (void)ws_size;
  const float* x     = (const float*)d_in[0];
  const float* g1f1w = (const float*)d_in[1];
  const float* g1f1b = (const float*)d_in[2];
  const float* g1f2w = (const float*)d_in[3];
  const float* g1f2b = (const float*)d_in[4];
  const float* g1gw  = (const float*)d_in[5];
  const float* g1gb  = (const float*)d_in[6];
  const float* g1lng = (const float*)d_in[7];
  const float* g1lnb = (const float*)d_in[8];
  const float* wq    = (const float*)d_in[9];
  const float* wk    = (const float*)d_in[10];
  const float* wv    = (const float*)d_in[11];
  const float* wow   = (const float*)d_in[12];
  const float* wob   = (const float*)d_in[13];
  const float* alng  = (const float*)d_in[14];
  const float* alnb  = (const float*)d_in[15];
  const float* g2f1w = (const float*)d_in[16];
  const float* g2f1b = (const float*)d_in[17];
  const float* g2f2w = (const float*)d_in[18];
  const float* g2f2b = (const float*)d_in[19];
  const float* g2gw  = (const float*)d_in[20];
  const float* g2gb  = (const float*)d_in[21];
  const float* g2lng = (const float*)d_in[22];
  const float* g2lnb = (const float*)d_in[23];
  const float* qhw   = (const float*)d_in[24];
  const float* qhb   = (const float*)d_in[25];
  float* out = (float*)d_out;

  char* ws = (char*)d_ws;
  size_t off = 0;
  auto alloc = [&](size_t bytes) -> void* {
    void* p = ws + off;
    off += (bytes + 255) & ~(size_t)255;
    return p;
  };
  const size_t WELEM = (size_t)D_MODEL * D_MODEL;
  bf16* wb_f1 = (bf16*)alloc(WELEM * 2);
  bf16* wb_f2 = (bf16*)alloc(WELEM * 2);
  bf16* wb_g  = (bf16*)alloc(WELEM * 2);
  bf16* wb_q  = (bf16*)alloc(WELEM * 2);
  bf16* wb_k  = (bf16*)alloc(WELEM * 2);
  bf16* wb_v  = (bf16*)alloc(WELEM * 2);
  bf16* wb_o  = (bf16*)alloc(WELEM * 2);
  float* eta1 = (float*)alloc((size_t)NTOK * D_MODEL * 4);
  float* eta2 = (float*)alloc((size_t)NTOK * D_MODEL * 4);
  float* hbuf = (float*)alloc((size_t)NTOK * D_MODEL * 4);
  float* ctx  = (float*)alloc((size_t)NTOK * D_MODEL * 4);
  float* h2   = (float*)alloc((size_t)NTOK * D_MODEL * 4);
  bf16* qb = (bf16*)alloc((size_t)NTOK * D_MODEL * 2);
  bf16* kb = (bf16*)alloc((size_t)NTOK * D_MODEL * 2);
  bf16* vb = (bf16*)alloc((size_t)NTOK * D_MODEL * 2);

  const dim3 blk(256);
  const int cvtGrid = (int)(WELEM / 256);
  f32_to_bf16_kernel<<<cvtGrid, blk, 0, stream>>>(g1f1w, wb_f1, (int)WELEM);
  f32_to_bf16_kernel<<<cvtGrid, blk, 0, stream>>>(g1f2w, wb_f2, (int)WELEM);
  f32_to_bf16_kernel<<<cvtGrid, blk, 0, stream>>>(g1gw,  wb_g,  (int)WELEM);
  f32_to_bf16_kernel<<<cvtGrid, blk, 0, stream>>>(wq,    wb_q,  (int)WELEM);
  f32_to_bf16_kernel<<<cvtGrid, blk, 0, stream>>>(wk,    wb_k,  (int)WELEM);
  f32_to_bf16_kernel<<<cvtGrid, blk, 0, stream>>>(wv,    wb_v,  (int)WELEM);
  f32_to_bf16_kernel<<<cvtGrid, blk, 0, stream>>>(wow,   wb_o,  (int)WELEM);

  const int gemmGrid = NTOK / 64;  // 128 blocks
  // GRN1
  gemm256_kernel<1, float><<<gemmGrid, blk, 0, stream>>>(
      x, wb_f1, g1f1b, nullptr, nullptr, nullptr, nullptr, eta1);
  gemm256_kernel<0, float><<<gemmGrid, blk, 0, stream>>>(
      eta1, wb_f2, g1f2b, nullptr, nullptr, nullptr, nullptr, eta2);
  gemm256_kernel<2, float><<<gemmGrid, blk, 0, stream>>>(
      eta2, wb_g, g1gb, x, eta2, g1lng, g1lnb, hbuf);
  // QKV projections (bf16 outputs)
  gemm256_kernel<0, bf16><<<gemmGrid, blk, 0, stream>>>(
      hbuf, wb_q, nullptr, nullptr, nullptr, nullptr, nullptr, qb);
  gemm256_kernel<0, bf16><<<gemmGrid, blk, 0, stream>>>(
      hbuf, wb_k, nullptr, nullptr, nullptr, nullptr, nullptr, kb);
  gemm256_kernel<0, bf16><<<gemmGrid, blk, 0, stream>>>(
      hbuf, wb_v, nullptr, nullptr, nullptr, nullptr, nullptr, vb);
  // flash attention
  attn_kernel<<<dim3(8 * NHEAD, TSEQ / 128), blk, 0, stream>>>(qb, kb, vb, ctx);
  // output projection + residual + LN
  gemm256_kernel<3, float><<<gemmGrid, blk, 0, stream>>>(
      ctx, wb_o, wob, hbuf, nullptr, alng, alnb, h2);
  // GRN2 + quantile heads on last timestep only (exact: GRN is pointwise)
  grn2_head_kernel<<<1, blk, 0, stream>>>(h2, g2f1w, g2f1b, g2f2w, g2f2b, g2gw,
                                          g2gb, g2lng, g2lnb, qhw, qhb, out);
}
